// paired_conv_1786706395588
// MI455X (gfx1250) — matrix operand order repaired and checked by reading, not compiled, not run on hardware
//
#include <hip/hip_runtime.h>

// CDNA5 wave32 WMMA types
typedef __attribute__((ext_vector_type(16))) _Float16 v16h;
typedef __attribute__((ext_vector_type(8)))  _Float16 v8h;
typedef __attribute__((ext_vector_type(8)))  float    v8f;

#define CN 64
#define HHH 128
#define WWD 128
#define HW (HHH * WWD)

#define PSTR 72                 // f16 per pixel slot (64 ch + 8 pad) -> 144B, 16B aligned
#define NCOL 130                // cols p=0..129  (ww = p-1, halo at p=0 and p=129)
#define ROWSTR (NCOL * PSTR)

// Folded weight: contribution of patch tap t=(i,j) channel c to output o=(u,v),
// before the center-subtract fold.
__device__ __forceinline__ float fold_w(int t, int c, int o,
                                        const float* __restrict__ w1,
                                        const float* __restrict__ w2,
                                        const float* __restrict__ w3) {
    const int i = t / 3, j = t % 3, u = o / 3, v = o % 3;
    const int du = i - u + 1, dv = j - v + 1;
    float a = 0.0f;
    if (du >= 0 && du < 3 && dv >= 0 && dv < 3) a += w3[c * 9 + du * 3 + dv];
    if (du >= 0 && du < 2 && dv >= 0 && dv < 2) a += w2[c * 4 + du * 2 + dv];
    if (du == 1 && dv == 1)                     a += w1[c];
    return a;
}

// Build B fragments (weights) in d_ws, pre-swizzled to this kernel's contiguous-k 16-bit
// B-matrix 32x16 layout: chunk q = tap*2 + half (K = 32 channels of one tap),
// lane L: N = L%16, K = (L<16?0:16) + 2r + {0,1} packed per dword r=0..7.
__global__ void pc_prep_weights(const float* __restrict__ w1,
                                const float* __restrict__ w2,
                                const float* __restrict__ w3,
                                unsigned int* __restrict__ wsB) {
    const int e = blockIdx.x * blockDim.x + threadIdx.x;
    if (e >= 18 * 32 * 8) return;
    const int q    = e >> 8;
    const int lane = (e >> 3) & 31;
    const int r    = e & 7;
    const int t = q >> 1, half = q & 1;
    const int o = lane & 15;
    const int kb = (lane < 16) ? 0 : 16;
    unsigned int packed = 0u;
    if (o < 9) {
        union { _Float16 h[2]; unsigned int u; } pk;
        #pragma unroll
        for (int s = 0; s < 2; ++s) {
            const int c = half * 32 + kb + 2 * r + s;
            float val = fold_w(t, c, o, w1, w2, w3);
            if (t == 4) {   // center-subtract fold: W'[4] = base[4] - sum_t base[t]
                float S = 0.0f;
                for (int tt = 0; tt < 9; ++tt) S += fold_w(tt, c, o, w1, w2, w3);
                val -= S;
            }
            pk.h[s] = (_Float16)val;
        }
        packed = pk.u;
    }
    wsB[(q * 32 + lane) * 8 + r] = packed;
}

// One block = one image row (n,h): 8 waves x 16-pixel strips.
// Stage x[h-1..h+1] into LDS as f16 [row][col][ch], then per wave run the
// 16x16x(576) GEMM as 18 x v_wmma_f32_16x16x32_f16 with 2x ds_load_b128 per
// A fragment and 2x ds_load_b128 per B fragment.
__global__ __launch_bounds__(256) void pc_main(
    const float* __restrict__ x,
    const float* __restrict__ b1, const float* __restrict__ b2,
    const float* __restrict__ b3,
    const unsigned int* __restrict__ wsB,
    float* __restrict__ out) {
    __shared__ __align__(16) _Float16     sx[3 * ROWSTR];   // 56,160 B
    __shared__ __align__(16) unsigned int sb[18 * 32 * 8];  // 18,432 B

    const int tid = threadIdx.x;
    const int blk = blockIdx.x;              // 0..511 = n*128 + h
    const int n = blk >> 7;
    const int h = blk & 127;
    const float* __restrict__ xn = x + (size_t)n * (CN * HW);

    // ---- copy pre-swizzled B fragments into LDS (18 u32 per thread) ----
    #pragma unroll
    for (int k = 0; k < 18; ++k) sb[tid + k * 256] = wsB[tid + k * 256];

    // ---- zero halo columns p=0 and p=129 (all 3 rows, full padded slot) ----
    {
        const int TOT = 3 * 2 * PSTR;        // 432
        for (int e = tid; e < TOT; e += 256) {
            const int i    = e / (2 * PSTR);
            const int rem  = e % (2 * PSTR);
            const int side = rem / PSTR;     // 0 -> p=0, 1 -> p=129
            const int c    = rem % PSTR;
            sx[(i * NCOL + side * 129) * PSTR + c] = (_Float16)0.0f;
        }
    }

    // ---- stage core: 3 rows x 64 ch x 32 float4 groups = 6144 items ----
    #pragma unroll 4
    for (int k = 0; k < 24; ++k) {
        const int e   = tid + k * 256;
        const int i   = e >> 11;             // / 2048
        const int c   = (e >> 5) & 63;
        const int g   = e & 31;
        const int hh  = h + i - 1;
        const int ww0 = g * 4;
        float4 v = {0.0f, 0.0f, 0.0f, 0.0f};
        if (hh >= 0 && hh < HHH)
            v = *(const float4*)(xn + (size_t)c * HW + hh * WWD + ww0);
        _Float16* dst = &sx[(i * NCOL + 1 + ww0) * PSTR + c];
        dst[0 * PSTR] = (_Float16)v.x;
        dst[1 * PSTR] = (_Float16)v.y;
        dst[2 * PSTR] = (_Float16)v.z;
        dst[3 * PSTR] = (_Float16)v.w;
    }
    __syncthreads();

    // ---- GEMM: M=16 pixels, N=16 (9 used), K=576 in 18 chunks of 32 ----
    const int lane  = tid & 31;
    const int wave  = tid >> 5;
    const int w0    = wave * 16;
    const int m     = lane & 15;             // A-matrix row (pixel in strip)
    const int abase = (lane < 16) ? 0 : 16;  // A-layout K sub-base
    const float Bsum = b1[0] + b2[0] + b3[0];

    v8f acc = {};
    #pragma unroll
    for (int t = 0; t < 9; ++t) {
        const int i = t / 3, j = t % 3;
        const int p = w0 + m + j;            // LDS col (ww = p-1), halo handles bounds
        const _Float16* base = &sx[(i * NCOL + p) * PSTR];
        #pragma unroll
        for (int half = 0; half < 2; ++half) {
            const int c0 = half * 32 + abase;
            const v8h lo = *(const v8h*)(base + c0);        // K_local +0..7
            const v8h hi = *(const v8h*)(base + c0 + 8);    // K_local +8..15
            const v16h a = __builtin_shufflevector(lo, hi,
                0, 1, 2, 3, 4, 5, 6, 7, 8, 9, 10, 11, 12, 13, 14, 15);
            const int q = t * 2 + half;
            const v16h b = ((const v16h*)sb)[q * 32 + lane];
            acc = __builtin_amdgcn_wmma_f32_16x16x32_f16(
                false, a, false, b, (short)0, acc, false, false);
        }
    }

    // ---- epilogue: C layout VGPR r -> (M = r + (lane<16?0:8), N = lane%16) ----
    const int o = lane & 15;
    if (o < 9) {
        const int mrow = (lane < 16) ? 0 : 8;
        const int P0 = blk * 128 + w0;       // n*16384 + h*128 + w0
        #pragma unroll
        for (int r = 0; r < 8; ++r) {
            const int P = P0 + mrow + r;
            out[P * 9 + o] = __expf(-(acc[r] + Bsum) * (1.0f / 27.0f));
        }
    }
}

extern "C" void kernel_launch(void* const* d_in, const int* in_sizes, int n_in,
                              void* d_out, int out_size, void* d_ws, size_t ws_size,
                              hipStream_t stream) {
    const float* x  = (const float*)d_in[0];
    const float* w1 = (const float*)d_in[1];
    const float* b1 = (const float*)d_in[2];
    const float* w2 = (const float*)d_in[3];
    const float* b2 = (const float*)d_in[4];
    const float* w3 = (const float*)d_in[5];
    const float* b3 = (const float*)d_in[6];
    unsigned int* wsB = (unsigned int*)d_ws;   // needs 18*32*8*4 = 18432 bytes
    float* out = (float*)d_out;

    // Fold the 3 convs + center-subtract into pre-swizzled WMMA B fragments.
    pc_prep_weights<<<18, 256, 0, stream>>>(w1, w2, w3, wsB);
    // One block per image row: 4*128 rows.
    pc_main<<<512, 256, 0, stream>>>(x, b1, b2, b3, wsB, out);
}
